// Transformer_43954695307591
// MI455X (gfx1250) — matrix operand order repaired and checked by reading, not compiled, not run on hardware
//
#include <hip/hip_runtime.h>
#include <hip/hip_bf16.h>
#include <math.h>

// Model dims (fixed by the reference)
#define LAYERS 8
#define SEQ    1024
#define HDIM   2048
#define NH     16
#define NKV    8
#define HD     128
#define IM     5632
#define VOCAB  32000
#define QKV_N  ((NH + 2 * NKV) * HD)   // 4096
#define EPSF   1e-5f
#define THETA  10000.0f

typedef __attribute__((ext_vector_type(16))) __bf16 v16bf;
typedef __attribute__((ext_vector_type(8)))  __bf16 v8bf;
typedef __attribute__((ext_vector_type(4)))  __bf16 v4bf;
typedef __attribute__((ext_vector_type(8)))  float  v8f;

static __device__ __forceinline__ __bf16 f2bf(float f) { return (__bf16)f; }

static __device__ __forceinline__ v16bf cat16(v8bf a, v8bf b) {
    return __builtin_shufflevector(a, b, 0, 1, 2, 3, 4, 5, 6, 7,
                                   8, 9, 10, 11, 12, 13, 14, 15);
}

// ---------------------------------------------------------------------------
// Embedding gather: x[s, :] = tok_emb[tokens[s], :]
// ---------------------------------------------------------------------------
__global__ __launch_bounds__(256) void embed_kernel(
    const int* __restrict__ tokens, const float* __restrict__ tok_emb,
    float* __restrict__ x)
{
    const int s = blockIdx.x;
    const int t = tokens[s];
    const float* src = tok_emb + (size_t)t * HDIM;
    float* dst = x + (size_t)s * HDIM;
    for (int i = threadIdx.x; i < HDIM; i += 256) dst[i] = src[i];
}

// ---------------------------------------------------------------------------
// Fused residual add + RMSNorm: ysum = a + b (b may be null => 0),
// out = rmsnorm(ysum) * w.  One block per row, H = 2048, 256 threads.
// ---------------------------------------------------------------------------
__global__ __launch_bounds__(256) void add_rmsnorm_kernel(
    const float* __restrict__ a, const float* __restrict__ b,
    const float* __restrict__ w, float* __restrict__ ysum,
    float* __restrict__ out)
{
    const int row = blockIdx.x;
    const float* ar = a + (size_t)row * HDIM;
    const float* br = b ? b + (size_t)row * HDIM : nullptr;
    float vals[HDIM / 256];
    float ss = 0.0f;
    for (int j = 0; j < HDIM / 256; ++j) {
        const int i = threadIdx.x + j * 256;
        float v = ar[i];
        if (br) v += br[i];
        vals[j] = v;
        ss += v * v;
    }
    for (int off = 16; off >= 1; off >>= 1) ss += __shfl_xor(ss, off, 32);
    __shared__ float red[8];
    if ((threadIdx.x & 31) == 0) red[threadIdx.x >> 5] = ss;
    __syncthreads();
    float tot = 0.0f;
    for (int wv = 0; wv < 8; ++wv) tot += red[wv];
    const float inv = rsqrtf(tot * (1.0f / HDIM) + EPSF);
    float* yr = ysum + (size_t)row * HDIM;
    float* orow = out + (size_t)row * HDIM;
    for (int j = 0; j < HDIM / 256; ++j) {
        const int i = threadIdx.x + j * 256;
        yr[i] = vals[j];
        orow[i] = vals[j] * inv * w[i];
    }
}

// ---------------------------------------------------------------------------
// Generic GEMM: C[M,N] = A[M,K] * B[K,N], fp32 in/out, bf16 WMMA compute.
// Block tile 128x128, BK=32, 256 threads = 8 waves (wave32).
// Waves arranged 4x2; each wave owns 32x64 = 2x4 WMMA 16x16 accumulators.
// A staged row-major, B staged TRANSPOSED so every WMMA fragment is two
// aligned 16-byte LDS vector loads (ds_load_b128) per lane.
// Requires M%128==0, N%128==0, K%32==0 (true for all call sites here).
// ---------------------------------------------------------------------------
#define BM 128
#define BN 128
#define BK 32
#define KPAD 8              // row stride 40 bf16 = 80 B (16-B aligned)

__global__ __launch_bounds__(256) void gemm_bf16_wmma(
    const float* __restrict__ A, const float* __restrict__ B,
    float* __restrict__ C, int M, int N, int K)
{
    __shared__ __align__(16) __bf16 lsA[BM][BK + KPAD];
    __shared__ __align__(16) __bf16 lsBt[BN][BK + KPAD];   // [col][k]
    const int bm = blockIdx.y * BM;
    const int bn = blockIdx.x * BN;
    const int tid = threadIdx.x;
    const int wave = tid >> 5;
    const int lane = tid & 31;
    const int waveM = wave >> 1;        // 0..3 -> 32-row strip
    const int waveN = wave & 1;         // 0..1 -> 64-col strip
    const int lhalf = lane & 15;
    const bool hi = lane >= 16;

    v8f acc[2][4];
    for (int r = 0; r < 2; ++r)
        for (int c = 0; c < 4; ++c)
            for (int i = 0; i < 8; ++i) acc[r][c][i] = 0.0f;

    // A staging: 2 threads per row, 16 cols each (two float4 -> one b128 store)
    const int arow = tid >> 1;
    const int acol = (tid & 1) << 4;
    // B staging: 4x4 sub-blocks; thread -> rows srow..+3, cols scol..+3
    const int scol = (tid & 31) << 2;   // 0..124
    const int srow = (tid >> 5) << 2;   // 0..28

    for (int k0 = 0; k0 < K; k0 += BK) {
        {
            const float4* ap = reinterpret_cast<const float4*>(
                A + (size_t)(bm + arow) * K + k0 + acol);
            float4 a0 = ap[0], a1 = ap[1], a2 = ap[2], a3 = ap[3];
            v8bf pk0, pk1;
            pk0[0] = f2bf(a0.x); pk0[1] = f2bf(a0.y);
            pk0[2] = f2bf(a0.z); pk0[3] = f2bf(a0.w);
            pk0[4] = f2bf(a1.x); pk0[5] = f2bf(a1.y);
            pk0[6] = f2bf(a1.z); pk0[7] = f2bf(a1.w);
            pk1[0] = f2bf(a2.x); pk1[1] = f2bf(a2.y);
            pk1[2] = f2bf(a2.z); pk1[3] = f2bf(a2.w);
            pk1[4] = f2bf(a3.x); pk1[5] = f2bf(a3.y);
            pk1[6] = f2bf(a3.z); pk1[7] = f2bf(a3.w);
            *reinterpret_cast<v8bf*>(&lsA[arow][acol])     = pk0;
            *reinterpret_cast<v8bf*>(&lsA[arow][acol + 8]) = pk1;

            // B: read 4 coalesced float4 rows, write 4 packed b64 columns
            float4 br_[4];
            for (int r = 0; r < 4; ++r)
                br_[r] = *reinterpret_cast<const float4*>(
                    B + (size_t)(k0 + srow + r) * N + bn + scol);
            for (int c = 0; c < 4; ++c) {
                const float e0 = c == 0 ? br_[0].x : c == 1 ? br_[0].y : c == 2 ? br_[0].z : br_[0].w;
                const float e1 = c == 0 ? br_[1].x : c == 1 ? br_[1].y : c == 2 ? br_[1].z : br_[1].w;
                const float e2 = c == 0 ? br_[2].x : c == 1 ? br_[2].y : c == 2 ? br_[2].z : br_[2].w;
                const float e3 = c == 0 ? br_[3].x : c == 1 ? br_[3].y : c == 2 ? br_[3].z : br_[3].w;
                v4bf pk;
                pk[0] = f2bf(e0); pk[1] = f2bf(e1);
                pk[2] = f2bf(e2); pk[3] = f2bf(e3);
                *reinterpret_cast<v4bf*>(&lsBt[scol + c][srow]) = pk;
            }
            if (k0 + BK < K) {   // prefetch next K tile (global_prefetch_b8)
                __builtin_prefetch(A + (size_t)(bm + arow) * K + k0 + BK + acol, 0, 1);
                __builtin_prefetch(B + (size_t)(k0 + BK + srow) * N + bn + scol, 0, 1);
            }
        }
        __syncthreads();

        // A fragment (16x32 bf16, ISA layout): lanes 0-15 K 0-7/16-23,
        // lanes 16-31 K 8-15/24-31 -> two aligned b128 loads.
        v16bf afrag[2], bfrag[4];
        const int ka = hi ? 8 : 0;
        for (int r = 0; r < 2; ++r) {
            const int rowi = waveM * 32 + r * 16 + lhalf;
            const v8bf lo = *reinterpret_cast<const v8bf*>(&lsA[rowi][ka]);
            const v8bf hi8 = *reinterpret_cast<const v8bf*>(&lsA[rowi][ka + 16]);
            afrag[r] = cat16(lo, hi8);
        }
        // B fragment (32x16 bf16): lane col N, same K order as A
        // (lanes 0-15: K 0-7/16-23, lanes 16-31: K 8-15/24-31) -> two b128 loads.
        const int kb = hi ? 8 : 0;
        for (int c = 0; c < 4; ++c) {
            const int coli = waveN * 64 + c * 16 + lhalf;
            const v8bf lo = *reinterpret_cast<const v8bf*>(&lsBt[coli][kb]);
            const v8bf hi8 = *reinterpret_cast<const v8bf*>(&lsBt[coli][kb + 16]);
            bfrag[c] = cat16(lo, hi8);
        }
        for (int r = 0; r < 2; ++r)
            for (int c = 0; c < 4; ++c)
                acc[r][c] = __builtin_amdgcn_wmma_f32_16x16x32_bf16(
                    false, afrag[r], false, bfrag[c], (short)0, acc[r][c],
                    false, false);
        __syncthreads();
    }

    // f32 C layout: lane l, vgpr i -> row i + (l>=16 ? 8:0), col l&15
    const int rbase = hi ? 8 : 0;
    for (int r = 0; r < 2; ++r)
        for (int c = 0; c < 4; ++c) {
            const int row0 = bm + waveM * 32 + r * 16 + rbase;
            const int col = bn + waveN * 64 + c * 16 + lhalf;
            for (int i = 0; i < 8; ++i)
                C[(size_t)(row0 + i) * N + col] = acc[r][c][i];
        }
}

// ---------------------------------------------------------------------------
// RoPE on q & k, split qkv row into qbuf / kbuf / vbuf. One block per token.
// ---------------------------------------------------------------------------
__global__ __launch_bounds__(256) void rope_split_kernel(
    const float* __restrict__ qkv, float* __restrict__ qb,
    float* __restrict__ kb, float* __restrict__ vb)
{
    const int s = blockIdx.x;
    const float* row = qkv + (size_t)s * QKV_N;
    const float pos = (float)s;
    for (int p = threadIdx.x; p < NH * (HD / 2); p += 256) {
        const int h = p >> 6, j = p & 63;
        const float inv = __powf(THETA, -(float)j * (2.0f / HD));
        float sn, cs;
        __sincosf(pos * inv, &sn, &cs);
        const float x1 = row[h * HD + j];
        const float x2 = row[h * HD + 64 + j];
        float* q = qb + (size_t)s * (NH * HD) + h * HD;
        q[j]      = x1 * cs - x2 * sn;
        q[j + 64] = x1 * sn + x2 * cs;
    }
    for (int p = threadIdx.x; p < NKV * (HD / 2); p += 256) {
        const int h = p >> 6, j = p & 63;
        const float inv = __powf(THETA, -(float)j * (2.0f / HD));
        float sn, cs;
        __sincosf(pos * inv, &sn, &cs);
        const float x1 = row[NH * HD + h * HD + j];
        const float x2 = row[NH * HD + h * HD + 64 + j];
        float* k = kb + (size_t)s * (NKV * HD) + h * HD;
        k[j]      = x1 * cs - x2 * sn;
        k[j + 64] = x1 * sn + x2 * cs;
    }
    for (int i = threadIdx.x; i < NKV * HD; i += 256)
        vb[(size_t)s * (NKV * HD) + i] = row[(NH + NKV) * HD + i];
}

// ---------------------------------------------------------------------------
// Flash attention, causal, GQA rep=2. One wave per (head, 16-query block).
// S = (Q*scale) Kt via WMMA, online softmax with width-16 shuffles,
// P re-packed through LDS to WMMA A layout, O += P V via WMMA.
// ---------------------------------------------------------------------------
__global__ __launch_bounds__(256) void flash_attn_kernel(
    const float* __restrict__ qb, const float* __restrict__ kb,
    const float* __restrict__ vb, float* __restrict__ attn)
{
    __shared__ __align__(16) __bf16 lsP[8][16][32 + KPAD];  // per-wave P staging
    const int wave = threadIdx.x >> 5;
    const int lane = threadIdx.x & 31;
    const int g = blockIdx.x * 8 + wave;
    const int head = g >> 6;            // 64 q-blocks per head
    const int qblk = g & 63;
    const int kvh = head >> 1;          // NH/NKV = 2
    const int lhalf = lane & 15;
    const bool hi = lane >= 16;
    const int rbase = hi ? 8 : 0;
    const float scale = 0.088388347648318447f;   // 1/sqrt(128)

    // Q fragments: 16 rows x 128 features -> 4 A-frags, scale folded in
    v16bf qf[4];
    {
        const int qrow = qblk * 16 + lhalf;
        const float* qp = qb + (size_t)qrow * (NH * HD) + head * HD;
        const int ka = hi ? 8 : 0;
        for (int f = 0; f < 4; ++f)
            for (int e = 0; e < 8; ++e) {
                qf[f][e]     = f2bf(qp[f * 32 + ka + e] * scale);
                qf[f][e + 8] = f2bf(qp[f * 32 + ka + 16 + e] * scale);
            }
    }

    v8f o[8];
    for (int t = 0; t < 8; ++t)
        for (int i = 0; i < 8; ++i) o[t][i] = 0.0f;
    float m[8], lsum[8];
    for (int i = 0; i < 8; ++i) { m[i] = -3.0e38f; lsum[i] = 0.0f; }

    const int qend = qblk * 16 + 15;
    for (int j = 0; j <= qend; j += 32) {
        // scores for keys [j, j+32): two 16x16 tiles
        v8f sacc[2];
        for (int t = 0; t < 2; ++t) {
            for (int i = 0; i < 8; ++i) sacc[t][i] = 0.0f;
            const int key = j + t * 16 + lhalf;
            const float* kp = kb + (size_t)key * (NKV * HD) + kvh * HD;
            const int kb0 = hi ? 8 : 0;
            for (int f = 0; f < 4; ++f) {
                v16bf bf_;
                for (int e = 0; e < 16; ++e) bf_[e] = f2bf(kp[f * 32 + kb0 + e + (e & 8)]);
                sacc[t] = __builtin_amdgcn_wmma_f32_16x16x32_bf16(
                    false, qf[f], false, bf_, (short)0, sacc[t], false, false);
            }
        }
        // causal mask + online softmax (row r lives in element i across 16 lanes)
        float corr[8];
        for (int i = 0; i < 8; ++i) {
            const int qi = qblk * 16 + rbase + i;
            if (j + lhalf > qi)      sacc[0][i] = -3.0e38f;
            if (j + 16 + lhalf > qi) sacc[1][i] = -3.0e38f;
            float v = fmaxf(sacc[0][i], sacc[1][i]);
            v = fmaxf(v, __shfl_xor(v, 1, 16));
            v = fmaxf(v, __shfl_xor(v, 2, 16));
            v = fmaxf(v, __shfl_xor(v, 4, 16));
            v = fmaxf(v, __shfl_xor(v, 8, 16));
            const float mnew = fmaxf(m[i], v);
            corr[i] = __expf(m[i] - mnew);
            m[i] = mnew;
            const float p0 = __expf(sacc[0][i] - mnew);
            const float p1 = __expf(sacc[1][i] - mnew);
            float rs = p0 + p1;
            rs += __shfl_xor(rs, 1, 16);
            rs += __shfl_xor(rs, 2, 16);
            rs += __shfl_xor(rs, 4, 16);
            rs += __shfl_xor(rs, 8, 16);
            lsum[i] = lsum[i] * corr[i] + rs;
            lsP[wave][rbase + i][lhalf]      = f2bf(p0);
            lsP[wave][rbase + i][16 + lhalf] = f2bf(p1);
        }
        for (int t = 0; t < 8; ++t)
            for (int i = 0; i < 8; ++i) o[t][i] *= corr[i];
        asm volatile("s_wait_dscnt 0" ::: "memory");   // P staged, wave-local
        // read P back in 16x32 A-fragment layout (two aligned b128 loads)
        v16bf pf;
        {
            const int kp0 = hi ? 8 : 0;
            const v8bf lo = *reinterpret_cast<const v8bf*>(&lsP[wave][lhalf][kp0]);
            const v8bf hi8 = *reinterpret_cast<const v8bf*>(&lsP[wave][lhalf][kp0 + 16]);
            pf = cat16(lo, hi8);
        }
        // O += P @ V  (8 feature tiles of 16)
        const int kv0 = hi ? 8 : 0;
        for (int t = 0; t < 8; ++t) {
            v16bf vf;
            for (int e = 0; e < 16; ++e)
                vf[e] = f2bf(vb[(size_t)(j + kv0 + e + (e & 8)) * (NKV * HD) +
                               kvh * HD + t * 16 + lhalf]);
            o[t] = __builtin_amdgcn_wmma_f32_16x16x32_bf16(
                false, pf, false, vf, (short)0, o[t], false, false);
        }
    }
    for (int t = 0; t < 8; ++t)
        for (int i = 0; i < 8; ++i) {
            const int r = qblk * 16 + rbase + i;
            attn[(size_t)r * (NH * HD) + head * HD + t * 16 + lhalf] =
                o[t][i] / lsum[i];
        }
}

// ---------------------------------------------------------------------------
// SwiGLU: out = silu(g) * u, g = gu[:, :IM], u = gu[:, IM:]
// ---------------------------------------------------------------------------
__global__ __launch_bounds__(256) void silu_mul_kernel(
    const float* __restrict__ gu, float* __restrict__ outv)
{
    const size_t idx = (size_t)blockIdx.x * 256 + threadIdx.x;
    const size_t srow = idx / IM;
    const size_t c = idx % IM;
    const float gg = gu[srow * (2 * IM) + c];
    const float uu = gu[srow * (2 * IM) + IM + c];
    outv[idx] = (gg / (1.0f + __expf(-gg))) * uu;
}

// ---------------------------------------------------------------------------
// Host orchestration
// ---------------------------------------------------------------------------
extern "C" void kernel_launch(void* const* d_in, const int* in_sizes, int n_in,
                              void* d_out, int out_size, void* d_ws, size_t ws_size,
                              hipStream_t stream) {
    (void)in_sizes; (void)n_in; (void)out_size; (void)ws_size;
    const int*   tokens  = (const int*)d_in[0];
    // d_in[1] = start_pos (0), d_in[2] = kv_cache (fully overwritten; unused)
    const float* tok_emb = (const float*)d_in[3];
    const float* wqkv    = (const float*)d_in[4];
    const float* wo      = (const float*)d_in[5];
    const float* wu      = (const float*)d_in[6];
    const float* w2      = (const float*)d_in[7];
    const float* anw     = (const float*)d_in[8];
    const float* fnw     = (const float*)d_in[9];
    const float* finw    = (const float*)d_in[10];
    const float* outw    = (const float*)d_in[11];
    float* out = (float*)d_out;

    float* ws = (float*)d_ws;
    size_t off = 0;
    const size_t SH = (size_t)SEQ * HDIM;
    float* bufA = ws + off; off += SH;                 // residual ping-pong
    float* bufB = ws + off; off += SH;
    float* bufC = ws + off; off += SH;
    float* bufD = ws + off; off += SH;
    float* y    = ws + off; off += SH;
    float* nrm  = ws + off; off += SH;
    float* nrm2 = ws + off; off += SH;
    float* qkv  = ws + off; off += (size_t)SEQ * QKV_N;
    float* qb   = ws + off; off += (size_t)SEQ * NH * HD;
    float* kbuf = ws + off; off += (size_t)SEQ * NKV * HD;
    float* vbuf = ws + off; off += (size_t)SEQ * NKV * HD;
    float* attn = ws + off; off += (size_t)SEQ * NH * HD;
    float* ao   = ws + off; off += SH;
    float* gu   = ws + off; off += (size_t)SEQ * 2 * IM;
    float* ffi  = ws + off; off += (size_t)SEQ * IM;

    embed_kernel<<<SEQ, 256, 0, stream>>>(tokens, tok_emb, bufA);

    float* xs0 = bufA; float* xs1 = bufB;   // (x, skip) pair
    float* hs0 = bufC; float* hs1 = bufD;   // (h, ffn) pair
    float* x = bufA; float* skip = nullptr; // layer 0: skip == 0

    const dim3 blk(256);
    for (int l = 0; l < LAYERS; ++l) {
        // y = x + skip; nrm = rmsnorm(y) * anw[l]
        add_rmsnorm_kernel<<<SEQ, blk, 0, stream>>>(x, skip, anw + (size_t)l * HDIM, y, nrm);
        // qkv = nrm @ wqkv[l]
        gemm_bf16_wmma<<<dim3(QKV_N / BN, SEQ / BM), blk, 0, stream>>>(
            nrm, wqkv + (size_t)l * HDIM * QKV_N, qkv, SEQ, QKV_N, HDIM);
        // RoPE + split
        rope_split_kernel<<<SEQ, blk, 0, stream>>>(qkv, qb, kbuf, vbuf);
        // attention
        flash_attn_kernel<<<NH * (SEQ / 16) / 8, blk, 0, stream>>>(qb, kbuf, vbuf, attn);
        // ao = attn @ wo[l]
        gemm_bf16_wmma<<<dim3(HDIM / BN, SEQ / BM), blk, 0, stream>>>(
            attn, wo + (size_t)l * (NH * HD) * HDIM, ao, SEQ, HDIM, NH * HD);
        // h = y + ao; nrm2 = rmsnorm(h) * fnw[l]
        add_rmsnorm_kernel<<<SEQ, blk, 0, stream>>>(y, ao, fnw + (size_t)l * HDIM, hs0, nrm2);
        // gu = nrm2 @ wu[l]
        gemm_bf16_wmma<<<dim3((2 * IM) / BN, SEQ / BM), blk, 0, stream>>>(
            nrm2, wu + (size_t)l * HDIM * (2 * IM), gu, SEQ, 2 * IM, HDIM);
        // ffi = silu(g) * u
        silu_mul_kernel<<<(SEQ * IM) / 256, blk, 0, stream>>>(gu, ffi);
        // ffn = ffi @ w2[l]
        gemm_bf16_wmma<<<dim3(HDIM / BN, SEQ / BM), blk, 0, stream>>>(
            ffi, w2 + (size_t)l * IM * HDIM, hs1, SEQ, HDIM, IM);
        // carry: (x, skip) <- (h, ffn); swap buffer pairs
        float* t0 = xs0; float* t1 = xs1;
        xs0 = hs0; xs1 = hs1;
        hs0 = t0;  hs1 = t1;
        x = xs0; skip = xs1;
    }

    // final = rmsnorm(h + ffn) * final_norm_w; out = final @ out_w
    add_rmsnorm_kernel<<<SEQ, blk, 0, stream>>>(x, skip, finw, y, nrm);
    gemm_bf16_wmma<<<dim3(VOCAB / BN, SEQ / BM), blk, 0, stream>>>(
        nrm, outw, out, SEQ, VOCAB, HDIM);
}
